// MambaBlock_11785390260480
// MI455X (gfx1250) — matrix operand order repaired and checked by reading, not compiled, not run on hardware
//
#include <hip/hip_runtime.h>
#include <stdint.h>

// ---------------- Mamba block on gfx1250 (wave32, WMMA bf16, async-LDS) -----
// Shapes: BSZ=2, T=2048, D=1024, E=2048, S=16, K=4.  M = BSZ*T = 4096.

#define BSZ 2
#define TT  2048
#define DD  1024
#define EE  2048
#define SS  16
#define KK  4
#define MM  (BSZ*TT)

typedef __attribute__((ext_vector_type(16))) __bf16 v16bf;
typedef __attribute__((ext_vector_type(8)))  float  v8f;

__device__ __forceinline__ unsigned short f2bf(float f) {
    union { float f; unsigned u; } a; a.f = f;
    unsigned u = a.u;
    unsigned r = u + 0x7FFFu + ((u >> 16) & 1u);   // round-to-nearest-even
    return (unsigned short)(r >> 16);
}

// async global->LDS copy of 16 bytes per lane (CDNA5, ASYNCcnt-tracked)
__device__ __forceinline__ void async_copy16(unsigned ldsOff, const void* g) {
    asm volatile("global_load_async_to_lds_b128 %0, %1, off"
                 :: "v"(ldsOff), "v"(g) : "memory");
}
#define WAIT_ASYNC_8() asm volatile("s_wait_asynccnt 0x8" ::: "memory")
#define WAIT_ASYNC_0() asm volatile("s_wait_asynccnt 0x0" ::: "memory")

// ------------------------------------------------ fp32 [K][N] -> bf16 [N][K]
__global__ __launch_bounds__(256)
void cvt_transpose_kernel(const float* __restrict__ in,
                          unsigned short* __restrict__ out, int Kd, int Nd) {
    __shared__ float tile[32][33];
    const int k0 = blockIdx.y * 32, n0 = blockIdx.x * 32;
    const int tx = threadIdx.x & 31, ty = threadIdx.x >> 5;  // 8 rows of 32
#pragma unroll
    for (int i = ty; i < 32; i += 8)
        tile[i][tx] = in[(size_t)(k0 + i) * Nd + n0 + tx];
    __syncthreads();
#pragma unroll
    for (int i = ty; i < 32; i += 8)
        out[(size_t)(n0 + i) * Kd + k0 + tx] = f2bf(tile[tx][i]);
}

// ---------------------------------------------------------------- LayerNorm
__global__ __launch_bounds__(256)
void ln_kernel(const float* __restrict__ x, const float* __restrict__ w,
               const float* __restrict__ b, unsigned short* __restrict__ out) {
    const int row = blockIdx.x;
    const int tid = threadIdx.x;
    const float* xr = x + (size_t)row * DD;
    float v[4];
    float s = 0.f, sq = 0.f;
#pragma unroll
    for (int i = 0; i < 4; ++i) {
        v[i] = xr[tid + 256 * i];
        s  += v[i];
        sq += v[i] * v[i];
    }
#pragma unroll
    for (int m = 16; m >= 1; m >>= 1) {
        s  += __shfl_xor(s,  m);
        sq += __shfl_xor(sq, m);
    }
    __shared__ float ss[8], ssq[8];
    __shared__ float mu_s, rstd_s;
    if ((tid & 31) == 0) { ss[tid >> 5] = s; ssq[tid >> 5] = sq; }
    __syncthreads();
    if (tid == 0) {
        float S = 0.f, Q = 0.f;
#pragma unroll
        for (int i = 0; i < 8; ++i) { S += ss[i]; Q += ssq[i]; }
        float mu  = S * (1.f / DD);
        float var = Q * (1.f / DD) - mu * mu;
        mu_s = mu;
        rstd_s = rsqrtf(var + 1e-5f);
    }
    __syncthreads();
    float mu = mu_s, r = rstd_s;
#pragma unroll
    for (int i = 0; i < 4; ++i) {
        int c = tid + 256 * i;
        out[(size_t)row * DD + c] = f2bf((v[i] - mu) * r * w[c] + b[c]);
    }
}

// ---------------------------------------------------------------- WMMA GEMM
// C[M,N] = epilogue( A[M,K](bf16, row-major) @ BT[N,K](bf16) + bias[N] )
// mode 0: plain   mode 1: exp()   mode 2: + resid[M,N]
// block tile 128x128, 8 waves (2x4), wave tile 64x32; K-step 64 (2 wmma steps)
// Double-buffered LDS filled by global_load_async_to_lds_b128.
#define KSTEP 64
#define LAS   72      // ushort row stride (144 B, 16B-aligned, bank-skewed)

union AFrag { unsigned u[8]; v16bf v; };

__global__ __launch_bounds__(256)
void gemm_bf16_kernel(const unsigned short* __restrict__ A,
                      const unsigned short* __restrict__ BT,
                      const float* __restrict__ bias,
                      const float* __restrict__ resid,
                      float* __restrict__ C,
                      int M, int N, int K, int mode) {
    __shared__ unsigned short lA[2][128 * LAS];   // [m][k]
    __shared__ unsigned short lB[2][128 * LAS];   // [n][k]

    const int tid  = threadIdx.x;
    const int lane = tid & 31;
    const int wave = tid >> 5;
    const int wm = wave >> 2;          // 0..1 -> 64-row half
    const int wn = wave & 3;           // 0..3 -> 32-col strip
    const int hl  = lane >> 4;         // lane half
    const int l16 = lane & 15;
    const int bm = blockIdx.y * 128;
    const int bn = blockIdx.x * 128;

    const unsigned aBase = (unsigned)(uintptr_t)(void*)&lA[0][0];
    const unsigned bBase = (unsigned)(uintptr_t)(void*)&lB[0][0];
    const unsigned bufBytes = 128 * LAS * 2;

    // stage one 128x64 A tile + 128x64 BT tile: 8 async b128 copies per thread
    auto stage = [&](int buf, int k0) {
        unsigned aB = aBase + (unsigned)buf * bufBytes;
        unsigned bB = bBase + (unsigned)buf * bufBytes;
#pragma unroll
        for (int c = tid; c < 1024; c += 256) {
            int r  = c >> 3;
            int kc = (c & 7) << 3;
            unsigned off = (unsigned)(r * LAS + kc) * 2u;
            async_copy16(aB + off, A  + (size_t)(bm + r) * K + k0 + kc);
            async_copy16(bB + off, BT + (size_t)(bn + r) * K + k0 + kc);
        }
    };

    v8f zero = {0.f,0.f,0.f,0.f,0.f,0.f,0.f,0.f};
    v8f acc[4][2];
#pragma unroll
    for (int i = 0; i < 4; ++i)
#pragma unroll
        for (int j = 0; j < 2; ++j) acc[i][j] = zero;

    const int nIter = K / KSTEP;
    stage(0, 0);
    for (int it = 0; it < nIter; ++it) {
        const int cur = it & 1;
        if (it + 1 < nIter) {
            stage(cur ^ 1, (it + 1) * KSTEP);
            WAIT_ASYNC_8();       // loads complete in order: current tile done
        } else {
            WAIT_ASYNC_0();
        }
        __syncthreads();

#pragma unroll
        for (int ks = 0; ks < 2; ++ks) {
            AFrag af[4], bf[2];
#pragma unroll
            for (int t = 0; t < 4; ++t) {
                int m = wm * 64 + t * 16 + l16;
                const unsigned short* base = &lA[cur][m * LAS + ks * 32];
#pragma unroll
                for (int i = 0; i < 8; ++i) {
                    int kb = ((i >> 2) << 4) + hl * 8 + ((i & 3) << 1);
                    af[t].u[i] = *(const unsigned*)&base[kb];
                }
            }
#pragma unroll
            for (int t = 0; t < 2; ++t) {
                int n = wn * 32 + t * 16 + l16;
                const unsigned short* base = &lB[cur][n * LAS + ks * 32];
#pragma unroll
                for (int j = 0; j < 8; ++j) {
                    int kb = ((j >> 2) << 4) + hl * 8 + ((j & 3) << 1);
                    bf[t].u[j] = *(const unsigned*)&base[kb];
                }
            }
#pragma unroll
            for (int tm = 0; tm < 4; ++tm)
#pragma unroll
                for (int tn = 0; tn < 2; ++tn)
                    acc[tm][tn] = __builtin_amdgcn_wmma_f32_16x16x32_bf16(
                        false, af[tm].v, false, bf[tn].v,
                        (short)0, acc[tm][tn], false, false);
        }
        __syncthreads();   // safe to overwrite 'cur' next iteration
    }

    // epilogue: C/D layout -> VGPR r: (m = r + 8*half, n = lane%16)
#pragma unroll
    for (int tm = 0; tm < 4; ++tm) {
#pragma unroll
        for (int tn = 0; tn < 2; ++tn) {
            int col = bn + wn * 32 + tn * 16 + l16;
            float bv = bias[col];
#pragma unroll
            for (int r = 0; r < 8; ++r) {
                int row = bm + wm * 64 + tm * 16 + r + hl * 8;
                float v = acc[tm][tn][r] + bv;
                if (mode == 1)      v = __expf(v);
                else if (mode == 2) v += resid[(size_t)row * N + col];
                C[(size_t)row * N + col] = v;
            }
        }
    }
}

// --------------------------------------------------- depthwise causal conv1d
__global__ __launch_bounds__(256)
void conv_kernel(const float* __restrict__ xp, const float* __restrict__ cw,
                 const float* __restrict__ cb,
                 float* __restrict__ xcv, unsigned short* __restrict__ xcvB) {
    const int e  = blockIdx.x * 256 + threadIdx.x;
    const int t0 = blockIdx.y * 256;
    const int b  = blockIdx.z;
    const float w0 = cw[e * KK + 0], w1 = cw[e * KK + 1];
    const float w2 = cw[e * KK + 2], w3 = cw[e * KK + 3];
    const float bb = cb[e];
    const float* src = xp + (size_t)b * TT * (2 * EE) + EE + e;
    float xm3 = (t0 >= 3) ? src[(size_t)(t0 - 3) * (2 * EE)] : 0.f;
    float xm2 = (t0 >= 2) ? src[(size_t)(t0 - 2) * (2 * EE)] : 0.f;
    float xm1 = (t0 >= 1) ? src[(size_t)(t0 - 1) * (2 * EE)] : 0.f;
#pragma unroll 4
    for (int t = t0; t < t0 + 256; ++t) {
        float xt = src[(size_t)t * (2 * EE)];
        float y  = w0 * xm3 + w1 * xm2 + w2 * xm1 + w3 * xt + bb;
        size_t idx = ((size_t)(b * TT + t)) * EE + e;
        xcv[idx]  = y;
        xcvB[idx] = f2bf(y);
        xm3 = xm2; xm2 = xm1; xm1 = xt;
    }
}

// ------------------------------------------------------------ selective scan
// 16 lanes per (b,e) channel, one state s per lane; 2 channels per wave.
__global__ __launch_bounds__(256)
void scan_kernel(const float* __restrict__ dtb, const float* __restrict__ xcv,
                 const float* __restrict__ xp,   // gate pre-activation cols [0,E)
                 const float* __restrict__ Aw, const float* __restrict__ Bw,
                 const float* __restrict__ Cw,
                 unsigned short* __restrict__ ycombB) {
    const int lane = threadIdx.x & 31;
    const int wave = threadIdx.x >> 5;
    const int hl   = lane >> 4;
    const int s    = lane & 15;
    const int c    = blockIdx.x * 16 + wave * 2 + hl;   // channel 0..B*E-1
    const int b    = c >> 11;                            // E = 2048
    const int e    = c & (EE - 1);

    const float Ae = Aw[e * SS + s];
    const float Be = Bw[e * SS + s];
    const float Ce = Cw[e * SS + s];
    float h = 0.f;

    const float* dts = dtb + (size_t)b * TT * EE + e;
    const float* xs  = xcv + (size_t)b * TT * EE + e;
    const float* gs  = xp  + (size_t)b * TT * (2 * EE) + e;
    unsigned short* ys = ycombB + (size_t)b * TT * EE + e;

    for (int t = 0; t < TT; ++t) {
        float dt = dts[(size_t)t * EE];
        float xt = xs[(size_t)t * EE];
        float a  = __expf(-dt * Ae);
        h = h * a + xt * Be;
        float p = h * Ce;
        p += __shfl_xor(p, 1);
        p += __shfl_xor(p, 2);
        p += __shfl_xor(p, 4);
        p += __shfl_xor(p, 8);
        if (s == 0) {
            float g  = gs[(size_t)t * (2 * EE)];
            float sg = 1.f / (1.f + __expf(-g));
            ys[(size_t)t * EE] = f2bf(p * sg);
        }
    }
}

// ----------------------------------------------------------------------------
extern "C" void kernel_launch(void* const* d_in, const int* in_sizes, int n_in,
                              void* d_out, int out_size, void* d_ws, size_t ws_size,
                              hipStream_t stream) {
    const float* x      = (const float*)d_in[0];
    const float* ln_w   = (const float*)d_in[1];
    const float* ln_b   = (const float*)d_in[2];
    const float* W_in   = (const float*)d_in[3];
    const float* b_in   = (const float*)d_in[4];
    const float* conv_w = (const float*)d_in[5];
    const float* conv_b = (const float*)d_in[6];
    const float* W_dt   = (const float*)d_in[7];
    const float* b_dt   = (const float*)d_in[8];
    const float* A      = (const float*)d_in[9];
    const float* B      = (const float*)d_in[10];
    const float* C      = (const float*)d_in[11];
    const float* W_out  = (const float*)d_in[12];
    const float* b_out  = (const float*)d_in[13];
    float* out = (float*)d_out;

    char* ws = (char*)d_ws;
    size_t off = 0;
    auto alloc = [&](size_t bytes) -> char* {
        char* p = ws + off;
        off = (off + bytes + 255) & ~(size_t)255;
        return p;
    };
    unsigned short* wInT   = (unsigned short*)alloc((size_t)DD * 2 * EE * 2); // [2E][D]
    unsigned short* wDtT   = (unsigned short*)alloc((size_t)EE * EE * 2);     // [E][E]
    unsigned short* wOutT  = (unsigned short*)alloc((size_t)EE * DD * 2);     // [D][E]
    unsigned short* xnB    = (unsigned short*)alloc((size_t)MM * DD * 2);
    float*          xpBuf  = (float*)alloc((size_t)MM * 2 * EE * 4);
    float*          xcv    = (float*)alloc((size_t)MM * EE * 4);
    unsigned short* xcvB   = (unsigned short*)alloc((size_t)MM * EE * 2);
    float*          dtb    = (float*)alloc((size_t)MM * EE * 4);
    unsigned short* ycombB = (unsigned short*)alloc((size_t)MM * EE * 2);
    (void)ws_size; (void)in_sizes; (void)n_in; (void)out_size;

    // 1) weights: fp32 [K][N] -> bf16 transposed [N][K]
    cvt_transpose_kernel<<<dim3(2 * EE / 32, DD / 32), 256, 0, stream>>>(
        W_in, wInT, DD, 2 * EE);
    cvt_transpose_kernel<<<dim3(EE / 32, EE / 32), 256, 0, stream>>>(
        W_dt, wDtT, EE, EE);
    cvt_transpose_kernel<<<dim3(DD / 32, EE / 32), 256, 0, stream>>>(
        W_out, wOutT, EE, DD);

    // 2) LayerNorm -> bf16
    ln_kernel<<<MM, 256, 0, stream>>>(x, ln_w, ln_b, xnB);

    // 3) in_proj: [4096,1024] @ [1024,4096] + b_in -> xp (fp32)
    gemm_bf16_kernel<<<dim3(2 * EE / 128, MM / 128), 256, 0, stream>>>(
        xnB, wInT, b_in, nullptr, xpBuf, MM, 2 * EE, DD, 0);

    // 4) depthwise causal conv over branch half of xp
    conv_kernel<<<dim3(EE / 256, TT / 256, BSZ), 256, 0, stream>>>(
        xpBuf, conv_w, conv_b, xcv, xcvB);

    // 5) dt = exp(xcv @ W_dt + b_dt)
    gemm_bf16_kernel<<<dim3(EE / 128, MM / 128), 256, 0, stream>>>(
        xcvB, wDtT, b_dt, nullptr, dtb, MM, EE, EE, 1);

    // 6) selective scan + sigmoid gating -> bf16
    scan_kernel<<<(BSZ * EE) / 16, 256, 0, stream>>>(
        dtb, xcv, xpBuf, A, B, C, ycombB);

    // 7) out_proj + bias + residual -> d_out (fp32)
    gemm_bf16_kernel<<<dim3(DD / 128, MM / 128), 256, 0, stream>>>(
        ycombB, wOutT, b_out, x, out, MM, DD, EE, 2);
}
